// SPGEMoeLayer_40415642255960
// MI455X (gfx1250) — matrix operand order repaired and checked by reading, not compiled, not run on hardware
//
#include <hip/hip_runtime.h>
#include <hip/hip_bf16.h>
#include <math.h>

// ---------------- problem constants (match reference) ----------------
#define TOK   8192      // T = B*S
#define HID   2048      // H
#define HSH   4096      // HS
#define PEXP  1024      // P
#define NEXP  8         // E

typedef __attribute__((ext_vector_type(16))) __bf16 v16bf;
typedef __attribute__((ext_vector_type(8)))  float  v8f;

union Frag {
    v16bf v;
    unsigned int u[8];
    uint4 q[2];
};

// float -> bf16 bits (round to nearest even) — used only in memory-bound
// conversion passes and scalar epilogues, never in GEMM inner loops.
__device__ __forceinline__ unsigned short f2bf(float f) {
    union { float f; unsigned u; } c; c.f = f;
    unsigned u = c.u;
    unsigned r = u + 0x7FFFu + ((u >> 16) & 1u);
    return (unsigned short)(r >> 16);
}
__device__ __forceinline__ unsigned pack2(float a, float b) {
    return (unsigned)f2bf(a) | ((unsigned)f2bf(b) << 16);
}
__device__ __forceinline__ float bflo(unsigned u) {
    union { unsigned u; float f; } c; c.u = u << 16; return c.f;
}
__device__ __forceinline__ float bfhi(unsigned u) {
    union { unsigned u; float f; } c; c.u = u & 0xffff0000u; return c.f;
}

// =====================================================================
// fp32 -> bf16 bulk conversion (memory bound; 8 elements / thread)
// =====================================================================
__global__ __launch_bounds__(256) void cvt_bf16_kernel(
    const float* __restrict__ in, unsigned short* __restrict__ out)
{
    const size_t i = ((size_t)blockIdx.x * 256 + threadIdx.x) * 8;
    const float4* p = reinterpret_cast<const float4*>(in + i);
    float4 a = p[0], b = p[1];
    uint4 o;
    o.x = pack2(a.x, a.y); o.y = pack2(a.z, a.w);
    o.z = pack2(b.x, b.y); o.w = pack2(b.z, b.w);
    *reinterpret_cast<uint4*>(out + i) = o;
}

// =====================================================================
// bf16 WMMA GEMM:  C[T, N] = A[T, K] * W^T  (+ epilogue)
//   MODE 1: W = sg_w  [N,K],  C = bf16, relu
//   MODE 2: W = sd_w  [N,K],  C = bf16
//   MODE 3: W = g_w   [N,K] (flat [E*P,H]), C = bf16, relu * wcomb[t, n>>10]
//   MODE 4: W = d_w   [E,H,P]: B[k,n] = d_w[k>>10, n, k&1023], C = fp32
// Block: 256 thr = 8 wave32 (2 M-groups x 4 N-groups), C tile 128x256.
// Wave tile 64x64 = 4x4 16x16 frags -> 16 WMMAs per K-step (K-step = 32).
// =====================================================================
template<int MODE, int N, int K>
__global__ __launch_bounds__(256) void gemm_wmma(
    const unsigned short* __restrict__ A,   // bf16 [T, K]
    const unsigned short* __restrict__ W,   // bf16 weights
    unsigned short* __restrict__ Cb,        // bf16 out (modes 1-3)
    float* __restrict__ Cf,                 // f32 out (mode 4)
    const float* __restrict__ scale)        // wcomb (mode 3)
{
    const int tid  = threadIdx.x;
    const int lane = tid & 31;
    const int wave = tid >> 5;
    const int lo   = lane & 15;
    const int hi   = lane >> 4;

    const int m0 = blockIdx.y * 128 + (wave & 1) * 64;   // 64 rows / wave
    const int n0 = blockIdx.x * 256 + (wave >> 1) * 64;  // 64 cols / wave

    v8f acc[4][4] = {};

    for (int k0 = 0; k0 < K; k0 += 32) {
        // A frags: lane -> M row; u[0..3] = K ka..ka+7, u[4..7] = K ka+16..ka+23
        Frag a[4];
#pragma unroll
        for (int mi = 0; mi < 4; ++mi) {
            const size_t base = (size_t)(m0 + mi * 16 + lo) * K + (k0 + 8 * hi);
            a[mi].q[0] = *reinterpret_cast<const uint4*>(A + base);
            a[mi].q[1] = *reinterpret_cast<const uint4*>(A + base + 16);
        }
        // B frags: lane -> N col; u[0..3] = K kb..kb+7, u[4..7] = K kb+16..kb+23
        Frag b[4];
        const int kb = k0 + 8 * hi;
#pragma unroll
        for (int ni = 0; ni < 4; ++ni) {
            const int col = n0 + ni * 16 + lo;
            const unsigned short* wr;
            if (MODE == 4) {
                // d_w [E,H,P]: two 8-elem runs in p, never cross p-block
                wr = W + ((size_t)(kb >> 10) * HID * PEXP) + (size_t)col * PEXP + (kb & 1023);
            } else {
                wr = W + (size_t)col * K + kb;
            }
            b[ni].q[0] = *reinterpret_cast<const uint4*>(wr);
            b[ni].q[1] = *reinterpret_cast<const uint4*>(wr + 16);
        }
        // 16 WMMAs per K-step per wave
#pragma unroll
        for (int mi = 0; mi < 4; ++mi)
#pragma unroll
            for (int ni = 0; ni < 4; ++ni)
                acc[mi][ni] = __builtin_amdgcn_wmma_f32_16x16x32_bf16(
                    false, a[mi].v, false, b[ni].v,
                    (short)0, acc[mi][ni], false, false);
    }

    // epilogue: lane(0-15) -> N col, hi half -> M+8, VGPR r -> M row
#pragma unroll
    for (int mi = 0; mi < 4; ++mi) {
#pragma unroll
        for (int ni = 0; ni < 4; ++ni) {
            const int colg = n0 + ni * 16 + lo;
#pragma unroll
            for (int r = 0; r < 8; ++r) {
                const int rowg = m0 + mi * 16 + hi * 8 + r;
                float v = acc[mi][ni][r];
                if (MODE == 1) {
                    Cb[(size_t)rowg * N + colg] = f2bf(fmaxf(v, 0.0f));
                } else if (MODE == 2) {
                    Cb[(size_t)rowg * N + colg] = f2bf(v);
                } else if (MODE == 3) {
                    v = fmaxf(v, 0.0f) * scale[(size_t)rowg * NEXP + (colg >> 10)];
                    Cb[(size_t)rowg * N + colg] = f2bf(v);
                } else {
                    Cf[(size_t)rowg * N + colg] = v;
                }
            }
        }
    }
}

// =====================================================================
// Router: wave per token. logits = x @ router_w.T ; softmax; top-2;
// combine weights to wcomb; importance/load stats via atomics.
// =====================================================================
__global__ __launch_bounds__(256) void router_kernel(
    const float* __restrict__ x, const float* __restrict__ rw,
    float* __restrict__ logits_out, float* __restrict__ wcomb,
    float* __restrict__ stats)
{
    const int lane = threadIdx.x & 31;
    const int wave = threadIdx.x >> 5;
    const int t = blockIdx.x * 8 + wave;

    float acc[NEXP] = {};
    const float* xr = x + (size_t)t * HID;
    for (int k = lane * 4; k < HID; k += 128) {
        float4 xv = *reinterpret_cast<const float4*>(xr + k);
#pragma unroll
        for (int e = 0; e < NEXP; ++e) {
            float4 rv = *reinterpret_cast<const float4*>(rw + (size_t)e * HID + k);
            acc[e] += xv.x * rv.x + xv.y * rv.y + xv.z * rv.z + xv.w * rv.w;
        }
    }
#pragma unroll
    for (int e = 0; e < NEXP; ++e)
#pragma unroll
        for (int off = 16; off; off >>= 1)
            acc[e] += __shfl_xor(acc[e], off, 32);

    if (lane == 0) {
        float mx = acc[0];
#pragma unroll
        for (int e = 1; e < NEXP; ++e) mx = fmaxf(mx, acc[e]);
        float p[NEXP], s = 0.0f;
#pragma unroll
        for (int e = 0; e < NEXP; ++e) { p[e] = __expf(acc[e] - mx); s += p[e]; }
        const float inv = 1.0f / s;
#pragma unroll
        for (int e = 0; e < NEXP; ++e) p[e] *= inv;

        int i1 = 0;
#pragma unroll
        for (int e = 1; e < NEXP; ++e) if (p[e] > p[i1]) i1 = e;
        int i2 = (i1 == 0) ? 1 : 0;
#pragma unroll
        for (int e = 0; e < NEXP; ++e) if (e != i1 && p[e] > p[i2]) i2 = e;

#pragma unroll
        for (int e = 0; e < NEXP; ++e) {
            logits_out[(size_t)t * NEXP + e] = acc[e];
            wcomb[(size_t)t * NEXP + e] = (e == i1) ? p[i1] : ((e == i2) ? p[i2] : 0.0f);
            atomicAdd(&stats[e], p[e]);
        }
        atomicAdd(&stats[NEXP + i1], 1.0f);
        atomicAdd(&stats[NEXP + i2], 1.0f);
    }
}

// =====================================================================
// Gate: wcomb[t,e] *= sigmoid(h1[t,:] @ pg_w[e,:])   (wave per token)
// =====================================================================
__global__ __launch_bounds__(256) void gate_kernel(
    const unsigned short* __restrict__ h1b, const float* __restrict__ pgw,
    float* __restrict__ wcomb)
{
    const int lane = threadIdx.x & 31;
    const int wave = threadIdx.x >> 5;
    const int t = blockIdx.x * 8 + wave;

    float acc[NEXP] = {};
    const unsigned short* hr = h1b + (size_t)t * HID;
    for (int k = lane * 8; k < HID; k += 256) {
        uint4 hv = *reinterpret_cast<const uint4*>(hr + k);
        float hf[8] = { bflo(hv.x), bfhi(hv.x), bflo(hv.y), bfhi(hv.y),
                        bflo(hv.z), bfhi(hv.z), bflo(hv.w), bfhi(hv.w) };
#pragma unroll
        for (int e = 0; e < NEXP; ++e) {
            const float4* pw = reinterpret_cast<const float4*>(pgw + (size_t)e * HID + k);
            float4 w0 = pw[0], w1 = pw[1];
            acc[e] += hf[0] * w0.x + hf[1] * w0.y + hf[2] * w0.z + hf[3] * w0.w
                    + hf[4] * w1.x + hf[5] * w1.y + hf[6] * w1.z + hf[7] * w1.w;
        }
    }
#pragma unroll
    for (int e = 0; e < NEXP; ++e)
#pragma unroll
        for (int off = 16; off; off >>= 1)
            acc[e] += __shfl_xor(acc[e], off, 32);

    if (lane == 0) {
#pragma unroll
        for (int e = 0; e < NEXP; ++e) {
            const float g = 1.0f / (1.0f + __expf(-acc[e]));
            wcomb[(size_t)t * NEXP + e] *= g;
        }
    }
}

__global__ void init_stats_kernel(float* stats) {
    if (threadIdx.x < 16) stats[threadIdx.x] = 0.0f;
}

__global__ void aux_loss_kernel(const float* __restrict__ stats, float* __restrict__ out) {
    float imps = 0.0f, lds = 0.0f;
#pragma unroll
    for (int e = 0; e < NEXP; ++e) { imps += stats[e]; lds += stats[NEXP + e]; }
    const float eps = 1e-9f;
    float aux = 0.0f;
#pragma unroll
    for (int e = 0; e < NEXP; ++e)
        aux += (stats[e] / (imps + eps)) * (stats[NEXP + e] / (lds + eps));
    out[0] = (float)NEXP * aux;
}

// =====================================================================
extern "C" void kernel_launch(void* const* d_in, const int* in_sizes, int n_in,
                              void* d_out, int out_size, void* d_ws, size_t ws_size,
                              hipStream_t stream)
{
    const float* x        = (const float*)d_in[0];  // [T, H]
    const float* router_w = (const float*)d_in[1];  // [E, H]
    const float* sg_w     = (const float*)d_in[2];  // [HS, H]
    const float* sd_w     = (const float*)d_in[3];  // [H, HS]
    const float* pg_w     = (const float*)d_in[4];  // [E, H]
    const float* g_w      = (const float*)d_in[5];  // [E, P, H] == flat [E*P, H]
    const float* d_w      = (const float*)d_in[6];  // [E, H, P]

    float* out_final  = (float*)d_out;                                   // T*H
    float* out_logits = out_final + (size_t)TOK * HID;                   // T*E
    float* out_aux    = out_logits + (size_t)TOK * NEXP;                 // 1

    // ---------------- workspace layout ----------------
    char* ws = (char*)d_ws;
    float* stats = (float*)ws;                                 // 16 f32
    float* wcomb = stats + 16;                                 // T*E f32
    size_t off = (16 + (size_t)TOK * NEXP) * sizeof(float);
    off = (off + 255) & ~(size_t)255;

    unsigned short* xb   = (unsigned short*)(ws + off); off += (size_t)TOK * HID * 2;   // x bf16
    unsigned short* sgb  = (unsigned short*)(ws + off); off += (size_t)HSH * HID * 2;   // sg_w bf16
    unsigned short* sdb  = (unsigned short*)(ws + off); off += (size_t)HID * HSH * 2;   // sd_w bf16
    unsigned short* gwb  = (unsigned short*)(ws + off); off += (size_t)NEXP * PEXP * HID * 2; // g_w bf16
    unsigned short* dwb  = (unsigned short*)(ws + off); off += (size_t)NEXP * HID * PEXP * 2; // d_w bf16
    unsigned short* mid  = (unsigned short*)(ws + off); off += (size_t)TOK * HSH * 2;   // [T,HS]
    unsigned short* h1   = (unsigned short*)(ws + off); off += (size_t)TOK * HID * 2;   // [T,H]
    unsigned short* aexp = (unsigned short*)(ws + off);                                  // [T,E*P]

    // ---------------- fp32 -> bf16 conversion passes ----------------
    cvt_bf16_kernel<<<(TOK * (size_t)HID) / (8 * 256), 256, 0, stream>>>(x, xb);
    cvt_bf16_kernel<<<((size_t)HSH * HID) / (8 * 256), 256, 0, stream>>>(sg_w, sgb);
    cvt_bf16_kernel<<<((size_t)HID * HSH) / (8 * 256), 256, 0, stream>>>(sd_w, sdb);
    cvt_bf16_kernel<<<((size_t)NEXP * PEXP * HID) / (8 * 256), 256, 0, stream>>>(g_w, gwb);
    cvt_bf16_kernel<<<((size_t)NEXP * HID * PEXP) / (8 * 256), 256, 0, stream>>>(d_w, dwb);

    // ---------------- router + aux stats ----------------
    init_stats_kernel<<<1, 32, 0, stream>>>(stats);
    router_kernel<<<TOK / 8, 256, 0, stream>>>(x, router_w, out_logits, wcomb, stats);
    aux_loss_kernel<<<1, 1, 0, stream>>>(stats, out_aux);

    // ---------------- GEMM chain (all WMMA bf16) ----------------
    // mid = relu(x @ sg_w.T)             [T, HS]
    gemm_wmma<1, HSH, HID><<<dim3(HSH / 256, TOK / 128), 256, 0, stream>>>(
        xb, sgb, mid, nullptr, nullptr);

    // h1 = mid @ sd_w.T                  [T, H]
    gemm_wmma<2, HID, HSH><<<dim3(HID / 256, TOK / 128), 256, 0, stream>>>(
        mid, sdb, h1, nullptr, nullptr);

    // wcomb[t,e] *= sigmoid(h1 @ pg_w.T)
    gate_kernel<<<TOK / 8, 256, 0, stream>>>(h1, pg_w, wcomb);

    // aexp[t, e*P+p] = relu(h1 @ g_w.T) * wcomb[t,e]     [T, E*P]
    gemm_wmma<3, NEXP * PEXP, HID><<<dim3((NEXP * PEXP) / 256, TOK / 128), 256, 0, stream>>>(
        h1, gwb, aexp, nullptr, wcomb);

    // final[t,h] = sum_k aexp[t,k] * d_w[k>>10, h, k&1023]   (K = E*P)
    gemm_wmma<4, HID, NEXP * PEXP><<<dim3(HID / 256, TOK / 128), 256, 0, stream>>>(
        aexp, dwb, nullptr, out_final, nullptr);
}
